// M2Voxelization_88012469830131
// MI455X (gfx1250) — matrix operand order repaired and checked by reading, not compiled, not run on hardware
//
#include <hip/hip_runtime.h>
#include <math.h>

#define RR 32
#define VV (RR*RR*RR)      // 32768 voxels
#define CC 240             // out channels
#define K2 480             // 2C contraction dim
#define BB 8
#define NN 16384
#define NT 15              // N tiles  (240/16)
#define KT 15              // K chunks (480/32)

typedef __attribute__((ext_vector_type(16))) _Float16 v16h;
typedef __attribute__((ext_vector_type(8)))  float    v8f;
typedef __attribute__((ext_vector_type(4)))  float    v4f;
typedef __attribute__((ext_vector_type(2)))  float    v2f;
typedef __attribute__((ext_vector_type(4)))  unsigned v4u;

// ---- monotonic float<->uint mapping for atomic max on f32 ----
__device__ __forceinline__ unsigned fmap(float f) {
  int i = __float_as_int(f);
  return (i >= 0) ? ((unsigned)i | 0x80000000u) : ~(unsigned)i;
}
__device__ __forceinline__ float funmap(unsigned u) {
  int i = (u & 0x80000000u) ? (int)(u & 0x7FFFFFFFu) : (int)~u;
  return __int_as_float(i);
}

// ---------------- coords -> nc + voxel idx (one block per batch) -------------
__global__ void norm_kernel(const float* __restrict__ coords,
                            float* __restrict__ nc_out,
                            int* __restrict__ idx) {
  const int b = blockIdx.x, tid = threadIdx.x;
  const float* cb = coords + (size_t)b * 3 * NN;
  __shared__ float shx[256], shy[256], shz[256];
  float sx = 0.f, sy = 0.f, sz = 0.f;
  for (int n = tid; n < NN; n += 256) {
    sx += cb[n]; sy += cb[NN + n]; sz += cb[2 * NN + n];
  }
  shx[tid] = sx; shy[tid] = sy; shz[tid] = sz;
  __syncthreads();
  for (int s = 128; s > 0; s >>= 1) {
    if (tid < s) { shx[tid] += shx[tid+s]; shy[tid] += shy[tid+s]; shz[tid] += shz[tid+s]; }
    __syncthreads();
  }
  const float mex = shx[0] * (1.f/NN), mey = shy[0] * (1.f/NN), mez = shz[0] * (1.f/NN);
  __syncthreads();
  float mxn = 0.f;
  for (int n = tid; n < NN; n += 256) {
    float dx = cb[n]-mex, dy = cb[NN+n]-mey, dz = cb[2*NN+n]-mez;
    mxn = fmaxf(mxn, sqrtf(dx*dx + dy*dy + dz*dz));
  }
  shx[tid] = mxn; __syncthreads();
  for (int s = 128; s > 0; s >>= 1) {
    if (tid < s) shx[tid] = fmaxf(shx[tid], shx[tid+s]);
    __syncthreads();
  }
  const float inv = 1.f / (shx[0] * 2.f);   // EPS == 0
  for (int n = tid; n < NN; n += 256) {
    float d0 = ((cb[n]        - mex) * inv + 0.5f) * (float)RR;
    float d1 = ((cb[NN + n]   - mey) * inv + 0.5f) * (float)RR;
    float d2 = ((cb[2*NN + n] - mez) * inv + 0.5f) * (float)RR;
    d0 = fminf(fmaxf(d0, 0.f), (float)(RR-1));
    d1 = fminf(fmaxf(d1, 0.f), (float)(RR-1));
    d2 = fminf(fmaxf(d2, 0.f), (float)(RR-1));
    nc_out[(size_t)b*3*NN + n]          = d0;
    nc_out[(size_t)b*3*NN + NN + n]     = d1;
    nc_out[(size_t)b*3*NN + 2*NN + n]   = d2;
    int vx = (int)rintf(d0), vy = (int)rintf(d1), vz = (int)rintf(d2); // RNE == jnp.round
    idx[(size_t)b*NN + n] = vx*(RR*RR) + vy*RR + vz;
  }
}

// ---------------- zero scratch for one batch (vectorized) -------------------
__global__ void init_kernel(v4u* __restrict__ sums4, v4u* __restrict__ mxu4,
                            float* __restrict__ cnt) {
  const size_t i = (size_t)blockIdx.x * 256 + threadIdx.x;  // over VV*CC/4
  const v4u z = {0u, 0u, 0u, 0u};
  const v4u m = {0x007FFFFFu, 0x007FFFFFu, 0x007FFFFFu, 0x007FFFFFu}; // fmap(-inf)
  if (i < (size_t)VV * CC / 4) { sums4[i] = z; mxu4[i] = m; }
  if (i < VV) cnt[i] = 0.f;
}

// ---------------- scatter sum + max + count (one batch) ---------------------
__global__ void scatter_kernel(const float* __restrict__ feat, const int* __restrict__ idx,
                               float* __restrict__ sums, unsigned* __restrict__ mxu,
                               float* __restrict__ cnt) {
  const int n = blockIdx.x * 256 + threadIdx.x;  // point (coalesced loads)
  const int c = blockIdx.y;                      // channel
  const int i = idx[n];
  const float f = feat[(size_t)c * NN + n];
  atomicAdd(&sums[(size_t)i * CC + c], f);
  atomicMax(&mxu[(size_t)i * CC + c], fmap(f));
  if (c == 0) atomicAdd(&cnt[i], 1.f);
}

// ---------------- pack conv_w into exact B-fragment order (f16) -------------
// B frag (16x16x32 f16): lane L holds col n=L%16; element e -> K=(e/8)*16+(L/16)*8+e%8.
__global__ void wpack_kernel(const float* __restrict__ w, _Float16* __restrict__ wpk) {
  const int id = blockIdx.x * 256 + threadIdx.x;
  if (id >= NT * KT * 32 * 16) return;
  const int e    = id & 15;
  const int lane = (id >> 4) & 31;
  const int kc   = (id >> 9) % KT;
  const int t    = id / (512 * KT);
  const int n = t * 16 + (lane & 15);
  const int k = kc * 32 + (e >> 3) * 16 + (lane >> 4) * 8 + (e & 7);
  wpk[id] = (_Float16)w[(size_t)n * K2 + k];
}

// ---------------- fold bias + BN into per-channel scale/shift ---------------
__global__ void prep_kernel(const float* __restrict__ bias, const float* __restrict__ gam,
                            const float* __restrict__ bet, const float* __restrict__ mu,
                            const float* __restrict__ var,
                            float* __restrict__ sc, float* __restrict__ sh) {
  const int o = blockIdx.x * 64 + threadIdx.x;
  if (o >= CC) return;
  const float s = gam[o] * rsqrtf(var[o] + 1e-5f);
  sc[o] = s;
  sh[o] = (bias[o] - mu[o]) * s + bet[o];
}

// ---------------- WMMA GEMM + scale/shift + swish (one batch) ---------------
// Wave owns one 16-row M tile x all 15 N tiles; K in 15 chunks of 32.
// Software-pipelined: B fragments double-buffered (loads for t+1 issued before
// wmma t), raw A vectors for kc+1 prefetched under the 15-wmma chain.
// Epilogue transposes 16x16 tiles via LDS -> full-cacheline b128 stores.
__global__ __launch_bounds__(128) void gemm_kernel(
    const unsigned* __restrict__ mxu, const float* __restrict__ sums,
    const float* __restrict__ cnt, const _Float16* __restrict__ wpk,
    const float* __restrict__ sc, const float* __restrict__ sh,
    float* __restrict__ out) {
  __shared__ float lds[4 * 5 * 16 * 20];               // 25.6 KB: 4 waves x 5 tiles x 16x20
  const int lane  = threadIdx.x;                       // 0..31
  const int mtile = blockIdx.x * 4 + threadIdx.y;      // 0..2047
  const int nn = lane & 15, h = lane >> 4;
  const size_t v = (size_t)mtile * 16 + nn;            // voxel row this lane reads for A
  const unsigned* rowmx = mxu  + v * CC;
  const float*  rowsum  = sums + v * CC;
  const float cn = cnt[v];
  const float inv = cn > 0.f ? 1.f / cn : 0.f;
  const bool occ = cn > 0.f;

  v8f acc[NT];
  const v8f zero = {0.f,0.f,0.f,0.f,0.f,0.f,0.f,0.f};
#pragma unroll
  for (int t = 0; t < NT; ++t) acc[t] = zero;

  // raw A loads: region (mx vs avg) is wave-uniform per (kc, r) since 240 % 16 == 0
  auto loadAraw = [&](int kc, v4u raw[4]) {
#pragma unroll
    for (int r = 0; r < 2; ++r) {
      const int ks = kc * 32 + r * 16 + h * 8;
      const unsigned* p = (kc * 32 + r * 16 < CC)
                              ? (rowmx + ks)
                              : (const unsigned*)(rowsum + (ks - CC));
      raw[2*r]     = *(const v4u*)p;
      raw[2*r + 1] = *(const v4u*)(p + 4);
    }
  };
  // A frag (16-bit A 16x32): element e=r*8+j -> K = kc*32 + r*16 + h*8 + j
  auto buildA = [&](int kc, const v4u raw[4]) -> v16h {
    v16h a;
#pragma unroll
    for (int r = 0; r < 2; ++r) {
      const bool isMx = (kc * 32 + r * 16) < CC;       // wave-uniform
#pragma unroll
      for (int p = 0; p < 2; ++p) {
        const v4u u = raw[2*r + p];
#pragma unroll
        for (int j = 0; j < 4; ++j) {
          float f;
          if (isMx) { f = funmap(u[j]); f = occ ? f : 0.f; }
          else      { f = __uint_as_float(u[j]) * inv; }
          a[r*8 + p*4 + j] = (_Float16)f;
        }
      }
    }
    return a;
  };

  v4u araw[4];
  loadAraw(0, araw);
#pragma unroll 1
  for (int kc = 0; kc < KT; ++kc) {
    const v16h a = buildA(kc, araw);
    v4u anext[4];
    if (kc + 1 < KT) loadAraw(kc + 1, anext);          // prefetch next K chunk
    // B double buffer: loads for tile t+1 issued before wmma for tile t
    v16h bf = *(const v16h*)(wpk + ((size_t)kc * 32 + lane) * 16);   // t = 0
#pragma unroll
    for (int t = 0; t < NT; ++t) {
      v16h bn = bf;
      if (t + 1 < NT)
        bn = *(const v16h*)(wpk + (((size_t)(t + 1) * KT + kc) * 32 + lane) * 16);
      acc[t] = __builtin_amdgcn_wmma_f32_16x16x32_f16(
          false, a, false, bf, (short)0, acc[t], false, false);
      bf = bn;
    }
#pragma unroll
    for (int r = 0; r < 4; ++r) araw[r] = anext[r];
  }

  // Epilogue: 3 stages of 5 tiles; LDS row stride 20 floats (16B-aligned quads)
  float* myl = lds + threadIdx.y * (5 * 16 * 20);
  const int row = (lane >> 2) & 7;
  const int q = lane & 3;
  const size_t v0 = (size_t)mtile * 16;
  for (int s = 0; s < 3; ++s) {
#pragma unroll
    for (int tl = 0; tl < 5; ++tl) {
      const int t = s * 5 + tl;
      const int o = t * 16 + nn;
      const float scv = sc[o], shv = sh[o];
      float* wrow = myl + (tl * 16 + nn) * 20 + h * 8;   // col = h*8 + r
#pragma unroll
      for (int r2 = 0; r2 < 4; ++r2) {
        float y0 = acc[t][2*r2]     * scv + shv;
        float y1 = acc[t][2*r2 + 1] * scv + shv;
        y0 = y0 / (1.f + __expf(-y0));                   // swish
        y1 = y1 / (1.f + __expf(-y1));
        v2f p = {y0, y1};
        *(v2f*)(wrow + 2 * r2) = p;                      // 8B-aligned ds_store_b64
      }
    }
    __syncthreads();
#pragma unroll
    for (int tl = 0; tl < 5; ++tl) {
#pragma unroll
      for (int g = 0; g < 2; ++g) {
        const int rr = g * 8 + row;
        const int o = (s * 5 + tl) * 16 + rr;
        const v4f val = *(const v4f*)(myl + (tl * 16 + rr) * 20 + q * 4);
        *(v4f*)(out + (size_t)o * VV + v0 + q * 4) = val; // 4 lanes = one full 64B line
      }
    }
    __syncthreads();
  }
}

extern "C" void kernel_launch(void* const* d_in, const int* in_sizes, int n_in,
                              void* d_out, int out_size, void* d_ws, size_t ws_size,
                              hipStream_t stream) {
  const float* features = (const float*)d_in[0];
  const float* coords   = (const float*)d_in[1];
  const float* conv_w   = (const float*)d_in[2];
  const float* conv_b   = (const float*)d_in[3];
  const float* gam      = (const float*)d_in[4];
  const float* bet      = (const float*)d_in[5];
  const float* bmean    = (const float*)d_in[6];
  const float* bvar     = (const float*)d_in[7];
  (void)in_sizes; (void)n_in; (void)out_size; (void)ws_size;

  float* out    = (float*)d_out;
  float* nc_out = out + (size_t)BB * CC * VV;   // second output, concatenated

  // workspace (~64 MB): sums | mxu | cnt | idx | wpack | sc | sh
  char* ws = (char*)d_ws;
  float*    sums = (float*)ws;                                         // VV*CC f32
  unsigned* mxu  = (unsigned*)(ws + (size_t)VV * CC * 4);              // VV*CC u32
  float*    cnt  = (float*)(ws + (size_t)VV * CC * 8);                 // VV f32
  int*      idx  = (int*)(ws + (size_t)VV * CC * 8 + (size_t)VV * 4);  // BB*NN i32
  _Float16* wpk  = (_Float16*)(ws + (size_t)VV * CC * 8 + (size_t)VV * 4
                                 + (size_t)BB * NN * 4);               // NT*KT*32*16 f16
  float*    scb  = (float*)((char*)wpk + (size_t)NT * KT * 32 * 16 * 2);
  float*    shb  = scb + CC;

  norm_kernel<<<BB, 256, 0, stream>>>(coords, nc_out, idx);
  wpack_kernel<<<(NT * KT * 32 * 16 + 255) / 256, 256, 0, stream>>>(conv_w, wpk);
  prep_kernel<<<(CC + 63) / 64, 64, 0, stream>>>(conv_b, gam, bet, bmean, bvar, scb, shb);

  for (int b = 0; b < BB; ++b) {
    init_kernel<<<(VV * CC / 4 + 255) / 256, 256, 0, stream>>>(
        (v4u*)sums, (v4u*)mxu, cnt);
    scatter_kernel<<<dim3(NN / 256, CC, 1), 256, 0, stream>>>(
        features + (size_t)b * CC * NN, idx + (size_t)b * NN, sums, mxu, cnt);
    gemm_kernel<<<dim3(VV / 16 / 4, 1, 1), dim3(32, 4, 1), 0, stream>>>(
        mxu, sums, cnt, wpk, scb, shb, out + (size_t)b * CC * VV);
  }
}
